// SlotAttention_20890720928590
// MI455X (gfx1250) — matrix operand order repaired and checked by reading, not compiled, not run on hardware
//
#include <hip/hip_runtime.h>
#include <hip/hip_bf16.h>

typedef __bf16 bf16;
typedef __attribute__((ext_vector_type(16))) __bf16 v16bf;
typedef __attribute__((ext_vector_type(8)))  float  v8f;

#define DEVINL static __device__ __forceinline__
#define V8F_ZERO {0.f,0.f,0.f,0.f,0.f,0.f,0.f,0.f}

DEVINL float wave_sum(float x) {
#pragma unroll
  for (int off = 16; off > 0; off >>= 1) x += __shfl_xor(x, off, 32);
  return x;
}

// A-fragment (16xK tile, row-major X, ld elems/row), per ISA 16-bit A layout:
// lane<16: m=lane, K = {k0..k0+7, k0+16..k0+23}; lane>=16: m=lane-16, K = {+8,+24}
DEVINL v16bf frag_a(const bf16* X, int ld, int m0, int k0, int lane) {
  int m = m0 + (lane & 15);
  int hi = lane >> 4;
  const bf16* p = X + (size_t)m * ld + k0 + hi * 8;
  union { uint4 u[2]; v16bf v; } t;
  t.u[0] = *(const uint4*)(p);
  t.u[1] = *(const uint4*)(p + 16);
  return t.v;
}

// B-fragment where B[k][n] = S[n][k] (S row-major, ld elems/row):
// lane: n = n0 + lane%16, K = {k0+(lane/16)*8 .. +7, k0+16+(lane/16)*8 .. +7}
DEVINL v16bf frag_b(const bf16* S, int ld, int n0, int k0, int lane) {
  int n = n0 + (lane & 15);
  int kh = lane >> 4;
  const bf16* p = S + (size_t)n * ld + k0 + kh * 8;
  union { uint4 u[2]; v16bf v; } t;
  t.u[0] = *(const uint4*)(p);
  t.u[1] = *(const uint4*)(p + 16);
  return t.v;
}

DEVINL v8f wmma_bf(v16bf a, v16bf b, v8f c) {
  return __builtin_amdgcn_wmma_f32_16x16x32_bf16(false, a, false, b, (short)0, c,
                                                 false, false);
}

// ---- CDNA5 async global->LDS copy (ASYNCcnt path, CDNA5 ISA) ------------
// VDST VGPR carries the LDS byte offset (low 32 bits of the generic LDS pointer),
// VADDR the 64-bit global address; GV addressing mode.
DEVINL void async_ld_b128(unsigned lds_off, const void* gptr) {
  asm volatile("global_load_async_to_lds_b128 %0, %1, off"
               :: "v"(lds_off), "v"((unsigned long long)(size_t)gptr)
               : "memory");
}
DEVINL void wait_async0() {
#if __has_builtin(__builtin_amdgcn_s_wait_asynccnt)
  __builtin_amdgcn_s_wait_asynccnt(0);
#else
  asm volatile("s_wait_asynccnt 0x0" ::: "memory");
#endif
}

DEVINL float sigmoidf(float x) { return 1.f / (1.f + __expf(-x)); }

// ---------------------------------------------------------------- convert f32->bf16
__global__ void k_convert(const float* __restrict__ src, bf16* __restrict__ dst, int n) {
  int i = blockIdx.x * 256 + threadIdx.x;
  if (i < n) dst[i] = (bf16)src[i];
}

// ---------------------------------------------------------------- LN of inp -> kv (bf16)
__global__ void k_ln_kv(const float* __restrict__ inp, const float* __restrict__ g,
                        const float* __restrict__ bb, bf16* __restrict__ out) {
  int wave = threadIdx.x >> 5, lane = threadIdx.x & 31;
  size_t row = (size_t)blockIdx.x * 8 + wave;
  const float4 x = *(const float4*)(inp + row * 128 + lane * 4);
  float s = wave_sum(x.x + x.y + x.z + x.w);
  float mean = s * (1.f / 128.f);
  float d0 = x.x - mean, d1 = x.y - mean, d2 = x.z - mean, d3 = x.w - mean;
  float v = wave_sum(d0 * d0 + d1 * d1 + d2 * d2 + d3 * d3);
  float inv = rsqrtf(v * (1.f / 128.f) + 1e-5f);
  const float4 gg = *(const float4*)(g + lane * 4);
  const float4 bv = *(const float4*)(bb + lane * 4);
  union { bf16 e[4]; uint2 u; } o;
  o.e[0] = (bf16)(d0 * inv * gg.x + bv.x);
  o.e[1] = (bf16)(d1 * inv * gg.y + bv.y);
  o.e[2] = (bf16)(d2 * inv * gg.z + bv.z);
  o.e[3] = (bf16)(d3 * inv * gg.w + bv.w);
  *(uint2*)(out + row * 128 + lane * 4) = o.u;
}

// ---------------------------------------------------------------- K = kv@Wk^T, Vt = (kv@Wv^T)^T
__global__ void k_proj(const bf16* __restrict__ kv, const bf16* __restrict__ Wk,
                       const bf16* __restrict__ Wv, bf16* __restrict__ kout,
                       bf16* __restrict__ vtout) {
  __shared__ __align__(16) bf16 at[16 * 128];   // staged A tile (4 KB)
  int wave = threadIdx.x >> 5, lane = threadIdx.x & 31;
  int m0 = blockIdx.x * 16;          // row tile of 131072 rows
  int which = blockIdx.y;            // 0=K, 1=V
  const bf16* W = which ? Wv : Wk;
  // async stage the 16x128 bf16 kv tile: 256 x b128
  async_ld_b128((unsigned)(size_t)at + threadIdx.x * 16,
                kv + (size_t)m0 * 128 + threadIdx.x * 8);
  wait_async0();
  __syncthreads();
  int n0 = wave * 16;
  v8f acc = V8F_ZERO;
#pragma unroll
  for (int kf = 0; kf < 4; ++kf) {
    v16bf a = frag_a(at, 128, 0, kf * 32, lane);
    v16bf b = frag_b(W, 128, n0, kf * 32, lane);
    acc = wmma_bf(a, b, acc);
  }
  int n = n0 + (lane & 15), hi = lane >> 4;
  if (!which) {
#pragma unroll
    for (int r = 0; r < 8; ++r) {
      size_t m = (size_t)(m0 + r + hi * 8);
      kout[m * 128 + n] = (bf16)acc[r];
    }
  } else {
#pragma unroll
    for (int r = 0; r < 8; ++r) {
      int gidx = m0 + r + hi * 8;
      int b_ = gidx >> 12, hw = gidx & 4095;     // HW = 4096
      vtout[((size_t)b_ * 128 + n) * 4096 + hw] = (bf16)acc[r];
    }
  }
}

// ---------------------------------------------------------------- qn = LN(q)@Wq^T * scale (bf16)
__global__ void k_qn(const float* __restrict__ q, const float* __restrict__ g,
                     const float* __restrict__ bb, const bf16* __restrict__ Wq,
                     bf16* __restrict__ qn) {
  __shared__ __align__(16) bf16 xbf[16 * 128];
  int b = blockIdx.x;
  int wave = threadIdx.x >> 5, lane = threadIdx.x & 31;
  const float* qb = q + (size_t)b * 16 * 128;
#pragma unroll
  for (int rr = 0; rr < 2; ++rr) {
    int row = wave * 2 + rr;
    const float4 x = *(const float4*)(qb + row * 128 + lane * 4);
    float s = wave_sum(x.x + x.y + x.z + x.w);
    float mean = s * (1.f / 128.f);
    float d0 = x.x - mean, d1 = x.y - mean, d2 = x.z - mean, d3 = x.w - mean;
    float v = wave_sum(d0 * d0 + d1 * d1 + d2 * d2 + d3 * d3);
    float inv = rsqrtf(v * (1.f / 128.f) + 1e-5f);
    const float4 gg = *(const float4*)(g + lane * 4);
    const float4 bv = *(const float4*)(bb + lane * 4);
    xbf[row * 128 + lane * 4 + 0] = (bf16)(d0 * inv * gg.x + bv.x);
    xbf[row * 128 + lane * 4 + 1] = (bf16)(d1 * inv * gg.y + bv.y);
    xbf[row * 128 + lane * 4 + 2] = (bf16)(d2 * inv * gg.z + bv.z);
    xbf[row * 128 + lane * 4 + 3] = (bf16)(d3 * inv * gg.w + bv.w);
  }
  __syncthreads();
  int n0 = wave * 16;
  v8f acc = V8F_ZERO;
#pragma unroll
  for (int kf = 0; kf < 4; ++kf) {
    v16bf a = frag_a(xbf, 128, 0, kf * 32, lane);
    v16bf bfr = frag_b(Wq, 128, n0, kf * 32, lane);
    acc = wmma_bf(a, bfr, acc);
  }
  const float scale = 0.08838834764831845f;   // 128^-0.5
  int n = n0 + (lane & 15), hi = lane >> 4;
#pragma unroll
  for (int r = 0; r < 8; ++r)
    qn[((size_t)b * 16 + r + hi * 8) * 128 + n] = (bf16)(acc[r] * scale);
}

// ---------------------------------------------------------------- zero slot sums + u accumulator
__global__ void k_zero(float* __restrict__ ssum, float* __restrict__ u) {
  int i = blockIdx.x * 256 + threadIdx.x;
  if (i < 65536) u[i] = 0.f;
  if (i < 512) ssum[i] = 0.f;
}

// ---------------------------------------------------------------- logits + softmax over slots
__global__ void k_logits(const bf16* __restrict__ qn, const bf16* __restrict__ kbf,
                         float* __restrict__ a0, float* __restrict__ ssum) {
  __shared__ __align__(16) bf16 qs[16 * 128];
  __shared__ float tile[8][16][17];
  int b = blockIdx.y, kg = blockIdx.x;
  int wave = threadIdx.x >> 5, lane = threadIdx.x & 31;
  // async stage this batch's qn (16x128 bf16 = 256 x b128)
  async_ld_b128((unsigned)(size_t)qs + threadIdx.x * 16,
                qn + (size_t)b * 2048 + threadIdx.x * 8);
  wait_async0();
  __syncthreads();
  int kpos0 = kg * 128 + wave * 16;
  const bf16* ks = kbf + (size_t)b * 4096 * 128;
  v8f acc = V8F_ZERO;
#pragma unroll
  for (int kf = 0; kf < 4; ++kf) {
    v16bf a = frag_a(qs, 128, 0, kf * 32, lane);
    v16bf bfr = frag_b(ks, 128, kpos0, kf * 32, lane);
    acc = wmma_bf(a, bfr, acc);
  }
  int n = lane & 15, hi = lane >> 4;
#pragma unroll
  for (int r = 0; r < 8; ++r) tile[wave][r + hi * 8][n] = acc[r];
  __syncthreads();
  if (lane < 16) {   // one key column per lane: softmax over the 16 slots
    float mx = -1e30f;
#pragma unroll
    for (int m = 0; m < 16; ++m) mx = fmaxf(mx, tile[wave][m][lane]);
    float e[16], s = 0.f;
#pragma unroll
    for (int m = 0; m < 16; ++m) { e[m] = __expf(tile[wave][m][lane] - mx); s += e[m]; }
    float inv = 1.f / s;
    int kp = kpos0 + lane;
#pragma unroll
    for (int m = 0; m < 16; ++m) {
      float v = e[m] * inv;
      tile[wave][m][lane] = v;
      a0[((size_t)b * 16 + m) * 4096 + kp] = v;
    }
  }
  __syncthreads();
  if (threadIdx.x < 16) {   // per-slot partial sum over this block's 128 key cols
    int m = threadIdx.x;
    float s = 0.f;
    for (int w = 0; w < 8; ++w)
#pragma unroll
      for (int c = 0; c < 16; ++c) s += tile[w][m][c];
    atomicAdd(&ssum[b * 16 + m], s);
  }
}

// ---------------------------------------------------------------- a = a0 / (sum_k + eps) -> bf16
__global__ void k_norm_a(const float* __restrict__ a0, const float* __restrict__ ssum,
                         bf16* __restrict__ abf) {
  size_t i = (size_t)blockIdx.x * 256 + threadIdx.x;
  int bs = (int)(i >> 12);                       // b*16 + slot
  abf[i] = (bf16)(a0[i] / (ssum[bs] + 1e-5f));
}

// ---------------------------------------------------------------- u = a @ v  (split-K, atomic f32)
__global__ void k_u(const bf16* __restrict__ abf, const bf16* __restrict__ vt,
                    float* __restrict__ u) {
  __shared__ __align__(16) bf16 at[16 * 512];    // 16 KB slice of `a`
  int b = blockIdx.y, ks = blockIdx.x;
  int wave = threadIdx.x >> 5, lane = threadIdx.x & 31;
  const bf16* A = abf + (size_t)b * 16 * 4096 + ks * 512;  // rows ld 4096
  const bf16* S = vt + (size_t)b * 128 * 4096;             // B-source: rows=c
  // async stage 16x512 bf16: 1024 x b128, 4 per thread
  unsigned ldsbase = (unsigned)(size_t)at;
#pragma unroll
  for (int i = 0; i < 4; ++i) {
    int t = threadIdx.x + i * 256;      // b128 index
    int row = t >> 6, c8 = t & 63;      // c8*8 bf16 within row
    async_ld_b128(ldsbase + (unsigned)(row * 1024 + c8 * 16),
                  A + (size_t)row * 4096 + c8 * 8);
  }
  wait_async0();
  __syncthreads();
  int n0 = wave * 16;
  int nrow = n0 + (lane & 15);
  v8f acc = V8F_ZERO;
#pragma unroll
  for (int kf = 0; kf < 16; ++kf) {
    int k0 = ks * 512 + kf * 32;
    if (kf < 14)   // prefetch B two chunks ahead (global_prefetch_b8)
      __builtin_prefetch(S + (size_t)nrow * 4096 + k0 + 64, 0, 1);
    v16bf a = frag_a(at, 512, 0, kf * 32, lane);
    v16bf bfr = frag_b(S, 4096, n0, k0, lane);
    acc = wmma_bf(a, bfr, acc);
  }
  int n = nrow, hi = lane >> 4;
#pragma unroll
  for (int r = 0; r < 8; ++r)
    atomicAdd(&u[((size_t)b * 16 + r + hi * 8) * 128 + n], acc[r]);
}

// ---------------------------------------------------------------- GRU + LN2 + FFN + residual
__global__ void k_gru_ffn(const float* __restrict__ u, const float* __restrict__ x,
                          const bf16* __restrict__ wih, const bf16* __restrict__ whh,
                          const float* __restrict__ bih, const float* __restrict__ bhh,
                          const float* __restrict__ g2, const float* __restrict__ b2ln,
                          const bf16* __restrict__ w1, const float* __restrict__ fb1,
                          const bf16* __restrict__ w2, const float* __restrict__ fb2,
                          float* __restrict__ qout) {
  __shared__ __align__(16) char smem[65536];
  bf16* ubf = (bf16*)smem;                    //  4 KB 16x128
  bf16* xbf = (bf16*)(smem + 4096);           //  4 KB 16x128
  float* gi = (float*)(smem + 8192);          // 24 KB 16x384
  float* gh = (float*)(smem + 32768);         // 24 KB 16x384
  float* yv = (float*)(smem + 57344);         //  8 KB 16x128
  bf16* zbf = (bf16*)smem;                    // reuse ubf after gates
  bf16* h1 = (bf16*)(smem + 8192);            // reuse gi after gates (16 KB 16x512)

  int wave = threadIdx.x >> 5, lane = threadIdx.x & 31;
  int row0 = blockIdx.x * 16;
  for (int idx = threadIdx.x; idx < 2048; idx += 256) {
    ubf[idx] = (bf16)u[(size_t)row0 * 128 + idx];
    xbf[idx] = (bf16)x[(size_t)row0 * 128 + idx];
  }
  __syncthreads();

  int nlo = lane & 15, hi = lane >> 4;
  // gi = u@wih^T + bih ; gh = x@whh^T + bhh  (24 col-tiles each, 3 per wave)
#pragma unroll
  for (int t = 0; t < 3; ++t) {
    int n0 = (wave * 3 + t) * 16;
    int n = n0 + nlo;
    v8f ai = V8F_ZERO, ah = V8F_ZERO;
#pragma unroll
    for (int kf = 0; kf < 4; ++kf) {
      v16bf bfr = frag_b(wih, 128, n0, kf * 32, lane);
      ai = wmma_bf(frag_a(ubf, 128, 0, kf * 32, lane), bfr, ai);
      v16bf bfr2 = frag_b(whh, 128, n0, kf * 32, lane);
      ah = wmma_bf(frag_a(xbf, 128, 0, kf * 32, lane), bfr2, ah);
    }
    float b_i = bih[n], b_h = bhh[n];
#pragma unroll
    for (int r = 0; r < 8; ++r) {
      gi[(r + hi * 8) * 384 + n] = ai[r] + b_i;
      gh[(r + hi * 8) * 384 + n] = ah[r] + b_h;
    }
  }
  __syncthreads();

  // gates -> y  (x reloaded in f32 for accuracy of the z*h path)
  for (int idx = threadIdx.x; idx < 2048; idx += 256) {
    int m = idx >> 7, c = idx & 127;
    float r = sigmoidf(gi[m * 384 + c] + gh[m * 384 + c]);
    float z = sigmoidf(gi[m * 384 + 128 + c] + gh[m * 384 + 128 + c]);
    float nn = tanhf(gi[m * 384 + 256 + c] + r * gh[m * 384 + 256 + c]);
    float xv = x[(size_t)row0 * 128 + idx];
    yv[idx] = (1.f - z) * nn + z * xv;
  }
  __syncthreads();

  // LN2(y) -> zbf
#pragma unroll
  for (int rr = 0; rr < 2; ++rr) {
    int row = wave * 2 + rr;
    const float4 xv = *(const float4*)(yv + row * 128 + lane * 4);
    float s = wave_sum(xv.x + xv.y + xv.z + xv.w);
    float mean = s * (1.f / 128.f);
    float d0 = xv.x - mean, d1 = xv.y - mean, d2 = xv.z - mean, d3 = xv.w - mean;
    float v = wave_sum(d0 * d0 + d1 * d1 + d2 * d2 + d3 * d3);
    float inv = rsqrtf(v * (1.f / 128.f) + 1e-5f);
    const float4 gg = *(const float4*)(g2 + lane * 4);
    const float4 bv = *(const float4*)(b2ln + lane * 4);
    zbf[row * 128 + lane * 4 + 0] = (bf16)(d0 * inv * gg.x + bv.x);
    zbf[row * 128 + lane * 4 + 1] = (bf16)(d1 * inv * gg.y + bv.y);
    zbf[row * 128 + lane * 4 + 2] = (bf16)(d2 * inv * gg.z + bv.z);
    zbf[row * 128 + lane * 4 + 3] = (bf16)(d3 * inv * gg.w + bv.w);
  }
  __syncthreads();

  // h1 = relu(z@w1^T + b1)   (32 col-tiles, 4 per wave)
#pragma unroll
  for (int t = 0; t < 4; ++t) {
    int n0 = (wave * 4 + t) * 16;
    int n = n0 + nlo;
    v8f acc = V8F_ZERO;
#pragma unroll
    for (int kf = 0; kf < 4; ++kf)
      acc = wmma_bf(frag_a(zbf, 128, 0, kf * 32, lane),
                    frag_b(w1, 128, n0, kf * 32, lane), acc);
    float bb = fb1[n];
#pragma unroll
    for (int r = 0; r < 8; ++r)
      h1[(r + hi * 8) * 512 + n] = (bf16)fmaxf(acc[r] + bb, 0.f);
  }
  __syncthreads();

  // q_out = y + h1@w2^T + b2  (8 col-tiles, 1 per wave; K=512)
  {
    int n0 = wave * 16;
    int n = n0 + nlo;
    v8f acc = V8F_ZERO;
#pragma unroll
    for (int kf = 0; kf < 16; ++kf)
      acc = wmma_bf(frag_a(h1, 512, 0, kf * 32, lane),
                    frag_b(w2, 512, n0, kf * 32, lane), acc);
    float bb = fb2[n];
#pragma unroll
    for (int r = 0; r < 8; ++r) {
      int m = r + hi * 8;
      qout[((size_t)row0 + m) * 128 + n] = yv[m * 128 + n] + acc[r] + bb;
    }
  }
}

// ---------------------------------------------------------------- final copy q -> d_out
__global__ void k_copy(const float* __restrict__ src, float* __restrict__ dst, int n) {
  int i = blockIdx.x * 256 + threadIdx.x;
  if (i < n) dst[i] = src[i];
}

// ================================================================ host
static size_t alignup(size_t x) { return (x + 255) & ~(size_t)255; }

extern "C" void kernel_launch(void* const* d_in, const int* in_sizes, int n_in,
                              void* d_out, int out_size, void* d_ws, size_t ws_size,
                              hipStream_t stream) {
  (void)in_sizes; (void)n_in; (void)out_size; (void)ws_size;
  const float* inp    = (const float*)d_in[0];
  const float* query  = (const float*)d_in[1];
  const float* lnkv_g = (const float*)d_in[2];
  const float* lnkv_b = (const float*)d_in[3];
  const float* Wk     = (const float*)d_in[4];
  const float* Wv     = (const float*)d_in[5];
  const float* lnq_g  = (const float*)d_in[6];
  const float* lnq_b  = (const float*)d_in[7];
  const float* Wq     = (const float*)d_in[8];
  const float* wih    = (const float*)d_in[9];
  const float* whh    = (const float*)d_in[10];
  const float* bih    = (const float*)d_in[11];
  const float* bhh    = (const float*)d_in[12];
  const float* ln2_g  = (const float*)d_in[13];
  const float* ln2_b  = (const float*)d_in[14];
  const float* w1     = (const float*)d_in[15];
  const float* fb1    = (const float*)d_in[16];
  const float* w2     = (const float*)d_in[17];
  const float* fb2    = (const float*)d_in[18];

  float* q_out  = (float*)d_out;            // 32*16*128
  float* a0_out = (float*)d_out + 65536;    // 32*16*4096

  // ---- workspace carve
  char* w = (char*)d_ws;
  size_t off = 0;
  auto take = [&](size_t bytes) { char* p = w + off; off = alignup(off + bytes); return p; };
  bf16* kv_bf  = (bf16*)take((size_t)16777216 * 2);  // LN'd inp
  bf16* k_bf   = (bf16*)take((size_t)16777216 * 2);  // K  [b][hw][c]
  bf16* vt_bf  = (bf16*)take((size_t)16777216 * 2);  // V^T [b][c][hw]
  bf16* a_bf   = (bf16*)take((size_t)2097152 * 2);   // renormalized attention
  bf16* qn_bf  = (bf16*)take((size_t)65536 * 2);
  float* u_f   = (float*)take((size_t)65536 * 4);
  float* qA    = (float*)take((size_t)65536 * 4);
  float* qB    = (float*)take((size_t)65536 * 4);
  float* ssum  = (float*)take((size_t)512 * 4);
  bf16* wq_bf  = (bf16*)take((size_t)16384 * 2);
  bf16* wk_bf  = (bf16*)take((size_t)16384 * 2);
  bf16* wv_bf  = (bf16*)take((size_t)16384 * 2);
  bf16* wih_bf = (bf16*)take((size_t)49152 * 2);
  bf16* whh_bf = (bf16*)take((size_t)49152 * 2);
  bf16* w1_bf  = (bf16*)take((size_t)65536 * 2);
  bf16* w2_bf  = (bf16*)take((size_t)65536 * 2);

  // ---- weight downconversion
  k_convert<<<64, 256, 0, stream>>>(Wq, wq_bf, 16384);
  k_convert<<<64, 256, 0, stream>>>(Wk, wk_bf, 16384);
  k_convert<<<64, 256, 0, stream>>>(Wv, wv_bf, 16384);
  k_convert<<<192, 256, 0, stream>>>(wih, wih_bf, 49152);
  k_convert<<<192, 256, 0, stream>>>(whh, whh_bf, 49152);
  k_convert<<<256, 256, 0, stream>>>(w1, w1_bf, 65536);
  k_convert<<<256, 256, 0, stream>>>(w2, w2_bf, 65536);

  // ---- kv = LN(inp); K, V^T projections (WMMA)
  k_ln_kv<<<16384, 256, 0, stream>>>(inp, lnkv_g, lnkv_b, kv_bf);
  k_proj<<<dim3(8192, 2), 256, 0, stream>>>(kv_bf, wk_bf, wv_bf, k_bf, vt_bf);

  // ---- iterative slot attention
  const float* qcur = query;
  float* bufs[2] = {qA, qB};
  for (int it = 0; it < 3; ++it) {
    float* qnext = bufs[it & 1];
    k_zero<<<256, 256, 0, stream>>>(ssum, u_f);
    k_qn<<<32, 256, 0, stream>>>(qcur, lnq_g, lnq_b, wq_bf, qn_bf);
    k_logits<<<dim3(32, 32), 256, 0, stream>>>(qn_bf, k_bf, a0_out, ssum);
    k_norm_a<<<8192, 256, 0, stream>>>(a0_out, ssum, a_bf);
    k_u<<<dim3(8, 32), 256, 0, stream>>>(a_bf, vt_bf, u_f);
    k_gru_ffn<<<32, 256, 0, stream>>>(u_f, qcur, wih_bf, whh_bf, bih, bhh,
                                      ln2_g, ln2_b, w1_bf, fb1, w2_bf, fb2, qnext);
    qcur = qnext;
  }
  k_copy<<<256, 256, 0, stream>>>(qcur, q_out, 65536);
}
